// MambaForLEDGAR_29111288333018
// MI455X (gfx1250) — matrix operand order repaired and checked by reading, not compiled, not run on hardware
//
#include <hip/hip_runtime.h>
#include <hip/hip_bf16.h>

#define B_      2
#define L_      1024
#define DM_     768
#define NLAYER_ 4
#define DI_     1536
#define DS_     16
#define DC_     4
#define DTR_    48
#define KPAD_   64                 // dt_proj K padded 48 -> 64
#define NLAB_   100
#define NTOK    (B_ * L_)          // 2048
#define XZW     (2 * DI_)          // 3072
#define XDBLW   (DTR_ + 2 * DS_)   // 80

typedef __bf16 bf16t;
typedef __attribute__((ext_vector_type(16))) bf16t v16bf;
typedef __attribute__((ext_vector_type(8)))  float v8f;

__device__ __forceinline__ unsigned short f2bf(float f) {
    unsigned int u = __float_as_uint(f);
    unsigned int r = u + 0x7FFFu + ((u >> 16) & 1u);   // round-to-nearest-even
    return (unsigned short)(r >> 16);
}

union Frag { v16bf v; uint4 q[2]; };

// ---------------------------------------------------------------------------
// C[M,N] = A[M,K] @ W[N,K]^T, bf16 in, f32 accumulate, 32x32 tile per wave
// (2x2 WMMA tiles). Requirements (guaranteed by the launcher):
//   M % 32 == 0, K % 32 == 0, N % 16 == 0, rows 16-byte aligned.
//   W rows up to ceil(N/32)*32 must be readable (scratch is oversized).
// mode 0: store; mode 1: += residual; mode 2: softplus(val + bias[n])
// ---------------------------------------------------------------------------
__global__ void gemm_wmma_bf16(const unsigned short* __restrict__ A, int lda,
                               const unsigned short* __restrict__ W,   // N x K row-major
                               float* __restrict__ C, int ldc,
                               int N, int K,
                               const float* __restrict__ residual,
                               const float* __restrict__ bias,
                               int mode)
{
    const int lane = threadIdx.x & 31;
    const int g = lane >> 4;           // half-wave group
    const int r = lane & 15;
    const int tile_n0 = blockIdx.x * 32;
    const int tile_m0 = blockIdx.y * 32;

    // per-lane row pointers (A rows: M index, W rows: N index, both K-contig)
    const unsigned short* __restrict__ a0row = A + (size_t)(tile_m0 + r)      * lda;
    const unsigned short* __restrict__ a1row = A + (size_t)(tile_m0 + 16 + r) * lda;
    const unsigned short* __restrict__ b0row = W + (size_t)(tile_n0 + r)      * K;
    const unsigned short* __restrict__ b1row = W + (size_t)(tile_n0 + 16 + r) * K;

    v8f acc00 = (v8f)(0.0f), acc01 = (v8f)(0.0f);
    v8f acc10 = (v8f)(0.0f), acc11 = (v8f)(0.0f);

    #pragma unroll 2
    for (int k0 = 0; k0 < K; k0 += 32) {
        if (k0 + 32 < K) {   // pull next weight slab (global_prefetch_b8)
            __builtin_prefetch(b0row + k0 + 32, 0, 1);
            __builtin_prefetch(b1row + k0 + 32, 0, 1);
        }
        Frag a0, a1, b0, b1;
        // A frag: lane holds K = {g*8 + 0..7} and {16 + g*8 + 0..7}
        a0.q[0] = *(const uint4*)(a0row + k0 + g * 8);
        a0.q[1] = *(const uint4*)(a0row + k0 + 16 + g * 8);
        a1.q[0] = *(const uint4*)(a1row + k0 + g * 8);
        a1.q[1] = *(const uint4*)(a1row + k0 + 16 + g * 8);
        // B frag: lane holds K = {g*8 + 0..7} and {16 + g*8 + 0..7}
        b0.q[0] = *(const uint4*)(b0row + k0 + g * 8);
        b0.q[1] = *(const uint4*)(b0row + k0 + 16 + g * 8);
        b1.q[0] = *(const uint4*)(b1row + k0 + g * 8);
        b1.q[1] = *(const uint4*)(b1row + k0 + 16 + g * 8);

        acc00 = __builtin_amdgcn_wmma_f32_16x16x32_bf16(false, a0.v, false, b0.v,
                                                        (short)0, acc00, false, false);
        acc10 = __builtin_amdgcn_wmma_f32_16x16x32_bf16(false, a1.v, false, b0.v,
                                                        (short)0, acc10, false, false);
        acc01 = __builtin_amdgcn_wmma_f32_16x16x32_bf16(false, a0.v, false, b1.v,
                                                        (short)0, acc01, false, false);
        acc11 = __builtin_amdgcn_wmma_f32_16x16x32_bf16(false, a1.v, false, b1.v,
                                                        (short)0, acc11, false, false);
    }

    // D layout: lane -> col n = r (+tile), rows m = g*8 + vi
    #pragma unroll
    for (int ni = 0; ni < 2; ++ni) {
        int tn = tile_n0 + ni * 16;
        if (tn >= N) continue;                 // wave-uniform guard (N % 16 == 0)
        int n = tn + r;
        #pragma unroll
        for (int mi = 0; mi < 2; ++mi) {
            int mbase = tile_m0 + mi * 16 + g * 8;
            #pragma unroll
            for (int vi = 0; vi < 8; ++vi) {
                float val;
                if (mi == 0) val = (ni == 0) ? acc00[vi] : acc01[vi];
                else         val = (ni == 0) ? acc10[vi] : acc11[vi];
                size_t idx = (size_t)(mbase + vi) * ldc + n;
                if (mode == 1) {
                    val += residual[idx];
                } else if (mode == 2) {
                    val += bias[n];
                    val = (val > 20.0f) ? val : log1pf(__expf(val));
                }
                C[idx] = val;
            }
        }
    }
}

// ---------------------------------------------------------------------------
__global__ void embed_kernel(const int* __restrict__ ids, const float* __restrict__ emb,
                             float* __restrict__ x)
{
    int idx = blockIdx.x * blockDim.x + threadIdx.x;
    if (idx >= NTOK * DM_) return;
    int t = idx / DM_, d = idx % DM_;
    x[idx] = emb[(size_t)ids[t] * DM_ + d];
}

__global__ void f32_to_bf16_kernel(unsigned short* __restrict__ dst,
                                   const float* __restrict__ src, int count)
{
    int idx = blockIdx.x * blockDim.x + threadIdx.x;
    if (idx < count) dst[idx] = f2bf(src[idx]);
}

// convert with column slicing + zero padding: dst[r, c] = c < take ? src[r*ld_src + c] : 0
__global__ void cvt_pad_kernel(unsigned short* __restrict__ dst,
                               const float* __restrict__ src,
                               int rows, int ld_src, int take, int dst_cols)
{
    int idx = blockIdx.x * blockDim.x + threadIdx.x;
    if (idx >= rows * dst_cols) return;
    int rr = idx / dst_cols, cc = idx % dst_cols;
    dst[idx] = (cc < take) ? f2bf(src[(size_t)rr * ld_src + cc]) : (unsigned short)0;
}

// causal depthwise conv (DC=4) + bias + SiLU; emits f32 and bf16 copies
__global__ void conv_silu_kernel(const float* __restrict__ xz, const float* __restrict__ cw,
                                 const float* __restrict__ cb, float* __restrict__ xc,
                                 unsigned short* __restrict__ xc_bf)
{
    int idx = blockIdx.x * blockDim.x + threadIdx.x;
    if (idx >= NTOK * DI_) return;
    int d = idx % DI_;
    int t = idx / DI_;
    int l = t % L_;
    int b = t / L_;
    float s = cb[d];
    #pragma unroll
    for (int j = 0; j < DC_; ++j) {
        int ls = l - (DC_ - 1) + j;
        if (ls >= 0)
            s += xz[(size_t)(b * L_ + ls) * XZW + d] * cw[d * DC_ + j];
    }
    float v = s * (1.0f / (1.0f + __expf(-s)));   // SiLU
    xc[idx] = v;
    xc_bf[idx] = f2bf(v);
}

// selective scan: one thread per (b, d) channel; DS=16 state in registers.
// Bm/Cm rows fetched as b128 vectors (16-byte aligned: 320-byte row stride,
// +192/+256 byte offsets) to minimize serial-chain memory instructions.
__global__ void scan_kernel(const float* __restrict__ delta, const float* __restrict__ xc,
                            const float* __restrict__ xz, const float* __restrict__ xdbl,
                            const float* __restrict__ A_log, const float* __restrict__ Dp,
                            unsigned short* __restrict__ y_bf)
{
    int tid = blockIdx.x * blockDim.x + threadIdx.x;
    if (tid >= B_ * DI_) return;
    int d = tid % DI_;
    int b = tid / DI_;

    alignas(16) float Alg[DS_];
    #pragma unroll
    for (int q = 0; q < 4; ++q)
        *(float4*)(Alg + 4 * q) = ((const float4*)(A_log + (size_t)d * DS_))[q];
    float Ar[DS_];
    #pragma unroll
    for (int nn = 0; nn < DS_; ++nn) Ar[nn] = -__expf(Alg[nn]);
    float Dd = Dp[d];
    float h[DS_];
    #pragma unroll
    for (int nn = 0; nn < DS_; ++nn) h[nn] = 0.0f;

    for (int l = 0; l < L_; ++l) {
        size_t t = (size_t)b * L_ + l;
        float dl  = delta[t * DI_ + d];
        float xcv = xc[t * DI_ + d];
        float zv  = xz[t * XZW + DI_ + d];
        float du  = dl * xcv;

        const float* __restrict__ row = xdbl + t * XDBLW;
        alignas(16) float bmv[DS_], cmv[DS_];
        #pragma unroll
        for (int q = 0; q < 4; ++q) {
            *(float4*)(bmv + 4 * q) = ((const float4*)(row + DTR_))[q];        // Bm
            *(float4*)(cmv + 4 * q) = ((const float4*)(row + DTR_ + DS_))[q];  // Cm
        }

        float yv = 0.0f;
        #pragma unroll
        for (int nn = 0; nn < DS_; ++nn) {
            float dA = __expf(dl * Ar[nn]);
            h[nn] = dA * h[nn] + du * bmv[nn];
            yv += h[nn] * cmv[nn];
        }
        yv += Dd * xcv;
        yv *= zv * (1.0f / (1.0f + __expf(-zv)));            // gate: SiLU(z)
        y_bf[t * DI_ + d] = f2bf(yv);
    }
}

__global__ void ln_kernel(const float* __restrict__ x, const float* __restrict__ sc,
                          const float* __restrict__ bi, float* __restrict__ xn)
{
    int t = blockIdx.x;
    const float* __restrict__ row = x + (size_t)t * DM_;
    __shared__ float sh[256];
    float s = 0.f, s2 = 0.f;
    for (int i = threadIdx.x; i < DM_; i += blockDim.x) {
        float v = row[i]; s += v; s2 += v * v;
    }
    sh[threadIdx.x] = s; __syncthreads();
    for (int st = 128; st > 0; st >>= 1) {
        if ((int)threadIdx.x < st) sh[threadIdx.x] += sh[threadIdx.x + st];
        __syncthreads();
    }
    float mu = sh[0] / DM_;
    __syncthreads();
    sh[threadIdx.x] = s2; __syncthreads();
    for (int st = 128; st > 0; st >>= 1) {
        if ((int)threadIdx.x < st) sh[threadIdx.x] += sh[threadIdx.x + st];
        __syncthreads();
    }
    float var = sh[0] / DM_ - mu * mu;
    float inv = rsqrtf(var + 1e-5f);
    for (int i = threadIdx.x; i < DM_; i += blockDim.x)
        xn[(size_t)t * DM_ + i] = (row[i] - mu) * inv * sc[i] + bi[i];
}

__global__ void pool_kernel(const float* __restrict__ xn, const int* __restrict__ mask,
                            float* __restrict__ pooled)
{
    int idx = blockIdx.x * blockDim.x + threadIdx.x;
    if (idx >= B_ * DM_) return;
    int d = idx % DM_, b = idx / DM_;
    float s = 0.f, ms = 0.f;
    for (int l = 0; l < L_; ++l) {
        float mk = (float)mask[b * L_ + l];
        s += xn[(size_t)(b * L_ + l) * DM_ + d] * mk;
        ms += mk;
    }
    pooled[idx] = s / ms;
}

__global__ void cls_kernel(const float* __restrict__ pooled, const float* __restrict__ w,
                           const float* __restrict__ bias, float* __restrict__ out)
{
    int idx = blockIdx.x * blockDim.x + threadIdx.x;
    if (idx >= B_ * NLAB_) return;
    int j = idx % NLAB_, b = idx / NLAB_;
    float s = bias[j];
    for (int k = 0; k < DM_; ++k) s += pooled[b * DM_ + k] * w[j * DM_ + k];
    out[idx] = s;
}

// ---------------------------------------------------------------------------
extern "C" void kernel_launch(void* const* d_in, const int* in_sizes, int n_in,
                              void* d_out, int out_size, void* d_ws, size_t ws_size,
                              hipStream_t stream)
{
    (void)in_sizes; (void)n_in; (void)out_size;
    const int*   ids   = (const int*)d_in[0];
    const int*   amask = (const int*)d_in[1];
    const float* emb   = (const float*)d_in[2];
    const float* inw   = (const float*)d_in[3];
    const float* cw    = (const float*)d_in[4];
    const float* cb    = (const float*)d_in[5];
    const float* xpw   = (const float*)d_in[6];
    const float* dtw   = (const float*)d_in[7];
    const float* dtb   = (const float*)d_in[8];
    const float* alog  = (const float*)d_in[9];
    const float* dpar  = (const float*)d_in[10];
    const float* outw  = (const float*)d_in[11];
    const float* nsc   = (const float*)d_in[12];
    const float* nbi   = (const float*)d_in[13];
    const float* clsw  = (const float*)d_in[14];
    const float* clsb  = (const float*)d_in[15];
    float* logits = (float*)d_out;

    char* ws = (char*)d_ws;
    size_t off = 0;
    auto alloc = [&](size_t bytes) -> char* {
        char* p = ws + off;
        off = (off + bytes + 255) & ~(size_t)255;
        return p;
    };
    float*          x_f32    = (float*)alloc((size_t)NTOK * DM_ * 4);
    unsigned short* x_bf     = (unsigned short*)alloc((size_t)NTOK * DM_ * 2);
    float*          xz       = (float*)alloc((size_t)NTOK * XZW * 4);
    float*          xc       = (float*)alloc((size_t)NTOK * DI_ * 4);
    unsigned short* xc_bf    = (unsigned short*)alloc((size_t)NTOK * DI_ * 2);
    float*          xdbl     = (float*)alloc((size_t)NTOK * XDBLW * 4);
    unsigned short* xdbl_pad = (unsigned short*)alloc((size_t)NTOK * KPAD_ * 2);
    float*          delta    = (float*)alloc((size_t)NTOK * DI_ * 4);
    unsigned short* y_bf     = (unsigned short*)alloc((size_t)NTOK * DI_ * 2);
    unsigned short* w_bf     = (unsigned short*)alloc((size_t)XZW * DM_ * 2);  // largest weight
    unsigned short* dtw_pad  = (unsigned short*)alloc((size_t)DI_ * KPAD_ * 2);
    float*          xn       = (float*)alloc((size_t)NTOK * DM_ * 4);
    float*          pooled   = (float*)alloc((size_t)B_ * DM_ * 4);
    if (off > ws_size) return;

    dim3 tb(256);
    auto cdiv = [](int a, int b) { return (a + b - 1) / b; };

    embed_kernel<<<cdiv(NTOK * DM_, 256), tb, 0, stream>>>(ids, emb, x_f32);

    for (int i = 0; i < NLAYER_; ++i) {
        const float* inw_i  = inw  + (size_t)i * XZW * DM_;
        const float* cw_i   = cw   + (size_t)i * DI_ * DC_;
        const float* cb_i   = cb   + (size_t)i * DI_;
        const float* xpw_i  = xpw  + (size_t)i * XDBLW * DI_;
        const float* dtw_i  = dtw  + (size_t)i * DI_ * DTR_;
        const float* dtb_i  = dtb  + (size_t)i * DI_;
        const float* alog_i = alog + (size_t)i * DI_ * DS_;
        const float* dpar_i = dpar + (size_t)i * DI_;
        const float* outw_i = outw + (size_t)i * DM_ * DI_;

        // in_proj: xz = x @ inw^T   (2048 x 3072, K=768)
        f32_to_bf16_kernel<<<cdiv(NTOK * DM_, 256), tb, 0, stream>>>(x_bf, x_f32, NTOK * DM_);
        f32_to_bf16_kernel<<<cdiv(XZW * DM_, 256), tb, 0, stream>>>(w_bf, inw_i, XZW * DM_);
        gemm_wmma_bf16<<<dim3(XZW / 32, NTOK / 32), dim3(32), 0, stream>>>(
            x_bf, DM_, w_bf, xz, XZW, XZW, DM_, nullptr, nullptr, 0);

        // causal depthwise conv + SiLU
        conv_silu_kernel<<<cdiv(NTOK * DI_, 256), tb, 0, stream>>>(xz, cw_i, cb_i, xc, xc_bf);

        // x_proj: x_dbl = xc @ xpw^T   (2048 x 80, K=1536); N=80 -> 3 N-blocks, store-guarded
        f32_to_bf16_kernel<<<cdiv(XDBLW * DI_, 256), tb, 0, stream>>>(w_bf, xpw_i, XDBLW * DI_);
        gemm_wmma_bf16<<<dim3(cdiv(XDBLW, 32), NTOK / 32), dim3(32), 0, stream>>>(
            xc_bf, DI_, w_bf, xdbl, XDBLW, XDBLW, DI_, nullptr, nullptr, 0);

        // dt_proj + softplus: delta = softplus(dt @ dtw^T + dtb), K padded 48 -> 64
        cvt_pad_kernel<<<cdiv(NTOK * KPAD_, 256), tb, 0, stream>>>(
            xdbl_pad, xdbl, NTOK, XDBLW, DTR_, KPAD_);
        cvt_pad_kernel<<<cdiv(DI_ * KPAD_, 256), tb, 0, stream>>>(
            dtw_pad, dtw_i, DI_, DTR_, DTR_, KPAD_);
        gemm_wmma_bf16<<<dim3(DI_ / 32, NTOK / 32), dim3(32), 0, stream>>>(
            xdbl_pad, KPAD_, dtw_pad, delta, DI_, DI_, KPAD_, nullptr, dtb_i, 2);

        // selective scan (fused D*xc + SiLU(z) gate) -> y (bf16)
        scan_kernel<<<cdiv(B_ * DI_, 256), tb, 0, stream>>>(
            delta, xc, xz, xdbl, alog_i, dpar_i, y_bf);

        // out_proj + residual: x = y @ outw^T + x   (2048 x 768, K=1536)
        f32_to_bf16_kernel<<<cdiv(DM_ * DI_, 256), tb, 0, stream>>>(w_bf, outw_i, DM_ * DI_);
        gemm_wmma_bf16<<<dim3(DM_ / 32, NTOK / 32), dim3(32), 0, stream>>>(
            y_bf, DI_, w_bf, x_f32, DM_, DM_, DI_, x_f32, nullptr, 1);
    }

    ln_kernel<<<NTOK, tb, 0, stream>>>(x_f32, nsc, nbi, xn);
    pool_kernel<<<cdiv(B_ * DM_, 256), tb, 0, stream>>>(xn, amask, pooled);
    cls_kernel<<<1, tb, 0, stream>>>(pooled, clsw, clsb, logits);
}
